// VQVAE_55276229100053
// MI455X (gfx1250) — matrix operand order repaired and checked by reading, not compiled, not run on hardware
//
#include <hip/hip_runtime.h>
#include <cstdint>
#include <cstddef>

typedef __bf16 bf16;
typedef __attribute__((ext_vector_type(16))) __bf16 v16bf;
typedef __attribute__((ext_vector_type(8)))  float  v8f;

struct __align__(16) U4 { unsigned x, y, z, w; };
union Frag { U4 q[2]; v16bf v; };
union B8  { U4 q; bf16 h[8]; };

// CDNA5 async global->LDS staging (ASYNCcnt-tracked), if the toolchain declares it.
#if defined(__AMDGCN__) && __has_builtin(__builtin_amdgcn_global_load_async_to_lds_b128) && \
    __has_builtin(__builtin_amdgcn_s_wait_asynccnt)
#define USE_ASYNC_LDS 1
#else
#define USE_ASYNC_LDS 0
#endif

#if USE_ASYNC_LDS
// Builtin expects pointers to 16-byte int vectors (per clang diagnostic):
//   arg0: AS(1) v4i*  (global source), arg1: AS(3) v4i* (LDS dest), imm offset, imm cpol
typedef int v4i_t __attribute__((vector_size(4 * sizeof(int))));
typedef __attribute__((address_space(1))) v4i_t* gas_ptr;
typedef __attribute__((address_space(3))) v4i_t* las_ptr;

__device__ __forceinline__ void async_copy16(const bf16* g, bf16* l)
{
  // flat LDS address low 32 bits == LDS byte offset (aperture rule), so the
  // integer-cast round trip produces valid AS(1)/AS(3) pointers.
  __builtin_amdgcn_global_load_async_to_lds_b128(
      (gas_ptr)(unsigned long long)g,
      (las_ptr)(unsigned int)(unsigned long long)l,
      0, 0);
}
#endif

// ---------------------------------------------------------------------------
// Generic implicit-GEMM conv / GEMM with bf16 WMMA, f32 accumulate.
//   out[M][Co] = gather(in)[M][Ktot] x wp[Co][Ktot]^T + bias[Co]
//   M = N*Ho*Wo pixels (NHWC input); every 32-wide K chunk lies inside one tap.
// Block: 256 thr = 8 waves (2 M x 4 N), tile 128x128, BK=32,
// double-buffered LDS with async loads (pipelined, s_wait_asynccnt),
// division-free incremental (tap, ci) K stream.
// zpad: >=16 bytes of zero-filled global memory for OOB rows (keeps EXEC
// uniform so per-wave ASYNCcnt stays uniform).
// ---------------------------------------------------------------------------
__global__ __launch_bounds__(256) void conv_gemm_wmma(
    const bf16* __restrict__ in, const bf16* __restrict__ wp,
    const float* __restrict__ bias, float* __restrict__ out,
    const bf16* __restrict__ zpad,
    int Hi, int Wi, int Ci, int Ho, int Wo, int Co,
    int stride, int pad, int KW, int Ktot)
{
  __shared__ __align__(16) bf16 As[2][128 * 32];   // [m][k]
  __shared__ __align__(16) bf16 Bs[2][128 * 32];   // [n][k] (transposed stage)

  const int t     = threadIdx.x;
  const int lane  = t & 31;
  const int waveM = (t >> 5) >> 2;   // 0..1
  const int waveN = (t >> 5) & 3;    // 0..3
  const int m0 = blockIdx.x * 128;
  const int n0 = blockIdx.y * 128;
  const int lm   = lane & 15;
  const int half = lane >> 4;

  // ---- A-load duty: 2 rows (64 apart), one 16B segment each ----
  const int segA = t & 3;
  int an[2], aoys[2], aoxs[2], aldst[2];
#pragma unroll
  for (int u = 0; u < 2; ++u) {
    int row = (t >> 2) + u * 64;
    aldst[u] = row * 32 + segA * 8;          // LDS element offset
    int m = m0 + row;
    int ox = m % Wo;
    int tmp = m / Wo;
    int oy = tmp % Ho;
    an[u]  = tmp / Ho;
    aoys[u] = oy * stride - pad;             // iy = aoys + ky
    aoxs[u] = ox * stride - pad;             // ix = aoxs + kx
  }
  // ---- B-load duty: 2 x 16B per thread ----
  const bf16* wb[2];
  int bldst[2];
#pragma unroll
  for (int u = 0; u < 2; ++u) {
    int task = t + u * 256;
    int row  = task >> 2;
    int seg  = task & 3;
    wb[u]    = wp + (size_t)(n0 + row) * Ktot + seg * 8;   // + k0 at issue
    bldst[u] = row * 32 + seg * 8;
  }

  v8f zero = {0.f,0.f,0.f,0.f,0.f,0.f,0.f,0.f};
  v8f acc[4][2];
#pragma unroll
  for (int i = 0; i < 4; ++i)
#pragma unroll
    for (int j = 0; j < 2; ++j) acc[i][j] = zero;

  const int NC = KW * KW * (Ci >> 5);        // number of 32-wide K chunks

  // incremental issue-state (uniform across threads; no divisions)
  int kyI = 0, kxI = 0, ciI = 0, k0I = 0;

  auto advance = [&]() {
    ciI += 32; k0I += 32;
    if (ciI == Ci) { ciI = 0; if (++kxI == KW) { kxI = 0; ++kyI; } }
  };

  auto stageA_addr = [&](int u) -> const bf16* {
    int iy = aoys[u] + kyI;
    int ix = aoxs[u] + kxI;
    bool v = (iy >= 0) & (iy < Hi) & (ix >= 0) & (ix < Wi);
    return v ? in + ((size_t)(an[u] * Hi + iy) * Wi + ix) * (size_t)Ci + ciI + segA * 8
             : zpad;
  };

  auto compute = [&](const bf16* Asb, const bf16* Bsb) {
    Frag a[4], b[2];
#pragma unroll
    for (int i = 0; i < 4; ++i) {
      int base = (waveM * 64 + i * 16 + lm) * 32 + half * 8;
      a[i].q[0] = *reinterpret_cast<const U4*>(&Asb[base]);       // K 0..7 / 8..15
      a[i].q[1] = *reinterpret_cast<const U4*>(&Asb[base + 16]);  // K 16..23 / 24..31
    }
#pragma unroll
    for (int j = 0; j < 2; ++j) {
      int base = (waveN * 32 + j * 16 + lm) * 32 + half * 8;
      b[j].q[0] = *reinterpret_cast<const U4*>(&Bsb[base]);
      b[j].q[1] = *reinterpret_cast<const U4*>(&Bsb[base + 16]);
    }
#pragma unroll
    for (int i = 0; i < 4; ++i)
#pragma unroll
      for (int j = 0; j < 2; ++j)
        acc[i][j] = __builtin_amdgcn_wmma_f32_16x16x32_bf16(
            false, a[i].v, false, b[j].v, (short)0, acc[i][j], false, false);
  };

#if USE_ASYNC_LDS
  auto issue = [&](int buf) {    // 4 async loads per thread -> uniform ASYNCcnt
#pragma unroll
    for (int u = 0; u < 2; ++u) async_copy16(stageA_addr(u), &As[buf][aldst[u]]);
#pragma unroll
    for (int u = 0; u < 2; ++u) async_copy16(wb[u] + k0I, &Bs[buf][bldst[u]]);
    advance();
  };

  issue(0);                                  // prologue: chunk 0 -> buffer 0
  for (int c = 0; c < NC; ++c) {
    int cur = c & 1;
    if (c + 1 < NC) {                        // pipeline: kick next, wait current
      issue(cur ^ 1);
      __builtin_amdgcn_s_wait_asynccnt(4);   // in-order => chunk c landed
    } else {
      __builtin_amdgcn_s_wait_asynccnt(0);
    }
    __syncthreads();                         // everyone's chunk c visible
    compute(As[cur], Bs[cur]);
    __syncthreads();                         // all reads done before next overwrite
  }
#else
  for (int c = 0; c < NC; ++c) {
    __syncthreads();
#pragma unroll
    for (int u = 0; u < 2; ++u) {
      U4 val = *reinterpret_cast<const U4*>(stageA_addr(u));
      *reinterpret_cast<U4*>(&As[0][aldst[u]]) = val;
    }
#pragma unroll
    for (int u = 0; u < 2; ++u) {
      U4 w = *reinterpret_cast<const U4*>(wb[u] + k0I);
      *reinterpret_cast<U4*>(&Bs[0][bldst[u]]) = w;
    }
    advance();
    __syncthreads();
    compute(As[0], Bs[0]);
  }
#endif

  // epilogue: D layout lane = N, vgpr r = M (r / 8+r by lane half)
#pragma unroll
  for (int j = 0; j < 2; ++j) {
    int col = n0 + waveN * 32 + j * 16 + lm;
    float bv = bias[col];
#pragma unroll
    for (int i = 0; i < 4; ++i) {
      int mrow = m0 + waveM * 64 + i * 16 + half * 8;
#pragma unroll
      for (int r = 0; r < 8; ++r)
        out[(size_t)(mrow + r) * Co + col] = acc[i][j][r] + bv;
    }
  }
}

// ---------------------------------------------------------------------------
// conv1: 3->128, k3 s2 p1, NCHW f32 in -> NHWC f32 raw out. K=27, VALU.
// ---------------------------------------------------------------------------
__global__ __launch_bounds__(256) void conv1_kernel(
    const float* __restrict__ x, const float* __restrict__ w1p,
    const float* __restrict__ b1, float* __restrict__ out)
{
  int t  = threadIdx.x;
  int co = t & 127;
  int pix = blockIdx.x * 2 + (t >> 7);  // 0..262143 over (n,oy,ox)
  int ox = pix & 63;
  int oy = (pix >> 6) & 63;
  int n  = pix >> 12;
  float acc = b1[co];
  for (int ky = 0; ky < 3; ++ky) {
    int iy = oy * 2 + ky - 1;
    if (iy < 0 || iy > 127) continue;
    for (int kx = 0; kx < 3; ++kx) {
      int ix = ox * 2 + kx - 1;
      if (ix < 0 || ix > 127) continue;
      for (int ci = 0; ci < 3; ++ci) {
        float xv = x[((size_t)(n * 3 + ci) * 128 + iy) * 128 + ix];
        acc = fmaf(xv, w1p[((ky * 3 + kx) * 3 + ci) * 128 + co], acc);
      }
    }
  }
  out[(size_t)pix * 128 + co] = acc;
}

// ---------------------------------------------------------------------------
// BatchNorm: deterministic two-stage reduction over pixels (NHWC raw f32).
// ---------------------------------------------------------------------------
__global__ __launch_bounds__(256) void bn_stats_kernel(
    const float* __restrict__ in, int C, int ppb,
    float* __restrict__ psum, float* __restrict__ psq)
{
  __shared__ float ss[256], sq[256];
  int t = threadIdx.x;
  int c = t % C;
  int sub = t / C;
  int step = 256 / C;     // 1 or 2
  int p0 = blockIdx.x * ppb;
  float s = 0.f, q = 0.f;
  for (int i = sub; i < ppb; i += step) {
    float v = in[(size_t)(p0 + i) * C + c];
    s += v; q += v * v;
  }
  ss[t] = s; sq[t] = q;
  __syncthreads();
  if (t < C) {
    if (step == 2) { s = ss[t] + ss[t + C]; q = sq[t] + sq[t + C]; }
    else           { s = ss[t];             q = sq[t]; }
    psum[(size_t)blockIdx.x * C + t] = s;
    psq [(size_t)blockIdx.x * C + t] = q;
  }
}

__global__ void bn_finalize_kernel(const float* __restrict__ psum,
                                   const float* __restrict__ psq,
                                   int nb, int C, float invP,
                                   const float* __restrict__ g,
                                   const float* __restrict__ be,
                                   float* __restrict__ ab)
{
  int c = threadIdx.x;
  float s = 0.f, q = 0.f;
  for (int b = 0; b < nb; ++b) { s += psum[(size_t)b * C + c]; q += psq[(size_t)b * C + c]; }
  float mean = s * invP;
  float var  = q * invP - mean * mean;
  float a = g[c] * rsqrtf(var + 1e-5f);
  ab[c]     = a;
  ab[C + c] = be[c] - mean * a;
}

__global__ __launch_bounds__(256) void bn_lrelu_bf16_kernel(
    const float* __restrict__ in, const float* __restrict__ ab, int C,
    bf16* __restrict__ out)
{
  size_t i = (size_t)blockIdx.x * 256 + threadIdx.x;
  int c = (int)(i % C);
  float v = in[i] * ab[c] + ab[C + c];
  v = v >= 0.f ? v : 0.01f * v;
  out[i] = (bf16)v;
}

// ---------------------------------------------------------------------------
// z_e NHWC f32 -> NCHW f32 (d_out), and f32 -> bf16 copy
// ---------------------------------------------------------------------------
__global__ __launch_bounds__(256) void ze_out_kernel(
    const float* __restrict__ ze, float* __restrict__ dout)
{
  int i = blockIdx.x * 256 + threadIdx.x;       // 16,777,216 = NCHW flat
  int w = i & 31, h = (i >> 5) & 31, c = (i >> 10) & 255, n = i >> 18;
  dout[i] = ze[((size_t)((n << 5) + h) * 32 + w) * 256 + c];
}

__global__ __launch_bounds__(256) void f32_to_bf16_kernel(
    const float* __restrict__ in, bf16* __restrict__ out)
{
  size_t i = (size_t)blockIdx.x * 256 + threadIdx.x;
  out[i] = (bf16)in[i];
}

// ---------------------------------------------------------------------------
// VQ: codebook norms, row argmin over scores = ||e||^2 - 2 x.e, gather + loss
// ---------------------------------------------------------------------------
__global__ void enorm_kernel(const float* __restrict__ cb, float* __restrict__ en)
{
  int k = blockIdx.x, lane = threadIdx.x;       // 32 lanes
  float s = 0.f;
  for (int u = 0; u < 8; ++u) {
    float v = cb[(size_t)k * 256 + u * 32 + lane];
    s += v * v;
  }
  for (int off = 16; off; off >>= 1) s += __shfl_down(s, off);
  if (lane == 0) en[k] = s;
}

__global__ __launch_bounds__(256) void argmin_kernel(
    const float* __restrict__ scores, int* __restrict__ idx)
{
  int lane = threadIdx.x & 31;
  int row = blockIdx.x * 8 + (threadIdx.x >> 5);
  const float* r = scores + (size_t)row * 512;
  float bv = 3.4e38f; int bi = 0x7fffffff;
  for (int j = lane; j < 512; j += 32) {
    float v = r[j];
    if (v < bv || (v == bv && j < bi)) { bv = v; bi = j; }
  }
  for (int off = 16; off; off >>= 1) {
    float ov = __shfl_down(bv, off);
    int   oi = __shfl_down(bi, off);
    if (ov < bv || (ov == bv && oi < bi)) { bv = ov; bi = oi; }
  }
  if (lane == 0) idx[row] = bi;
}

__global__ __launch_bounds__(256) void vq_gather_kernel(
    const int* __restrict__ idx, const float* __restrict__ cb,
    const float* __restrict__ ze, float* __restrict__ dout_zq,
    bf16* __restrict__ zqbf, float* __restrict__ lossPart)
{
  __shared__ float red[256];
  int m = blockIdx.x;          // pixel
  int c = threadIdx.x;
  int k = idx[m];
  float q = cb[(size_t)k * 256 + c];
  float z = ze[(size_t)m * 256 + c];
  float d = q - z;
  zqbf[(size_t)m * 256 + c] = (bf16)q;
  int w = m & 31, h = (m >> 5) & 31, n = m >> 10;
  dout_zq[((size_t)((n * 256 + c) * 32 + h) << 5) + w] = q;
  red[c] = d * d;
  __syncthreads();
  for (int s = 128; s; s >>= 1) { if (c < s) red[c] += red[c + s]; __syncthreads(); }
  if (c == 0) lossPart[m] = red[0];
}

__global__ void loss_final_kernel(const float* __restrict__ part, float* __restrict__ out)
{
  __shared__ float red[256];
  float s = 0.f;
  for (int i = threadIdx.x; i < 65536; i += 256) s += part[i];
  red[threadIdx.x] = s;
  __syncthreads();
  for (int k = 128; k; k >>= 1) { if (threadIdx.x < k) red[threadIdx.x] += red[threadIdx.x + k]; __syncthreads(); }
  if (threadIdx.x == 0) out[0] = 1.25f * red[0] / 16777216.f;
}

// ---------------------------------------------------------------------------
// scatter z_q bf16 into zero-dilated (64,66,66,256) buffer for convT1-as-conv
// ---------------------------------------------------------------------------
__global__ __launch_bounds__(256) void scatter_dil_kernel(
    const bf16* __restrict__ zq, bf16* __restrict__ dil)
{
  int m = blockIdx.x;
  int c = threadIdx.x;
  int ix = m & 31, iy = (m >> 5) & 31, n = m >> 10;
  dil[((size_t)(n * 66 + 1 + 2 * iy) * 66 + (1 + 2 * ix)) * 256 + c] = zq[(size_t)m * 256 + c];
}

// ---------------------------------------------------------------------------
// convT2: 256->3 k3 s2 p1 op1, gather form + tanh, wave-per-pixel over Cin
// ---------------------------------------------------------------------------
__global__ __launch_bounds__(256) void deconv2_tanh_kernel(
    const bf16* __restrict__ d1, const float* __restrict__ wp,
    const float* __restrict__ bd2, float* __restrict__ out)
{
  int lane = threadIdx.x & 31;
  int pix = blockIdx.x * 8 + (threadIdx.x >> 5);  // 0..1048575 over (n,oy,ox)
  int ox = pix & 127, oy = (pix >> 7) & 127, n = pix >> 14;
  float a0 = 0.f, a1 = 0.f, a2 = 0.f;
  for (int ky = 0; ky < 3; ++ky) {
    int ty = oy + 1 - ky;
    if (ty & 1) continue;
    int iy = ty >> 1;
    if (iy < 0 || iy >= 64) continue;
    for (int kx = 0; kx < 3; ++kx) {
      int tx = ox + 1 - kx;
      if (tx & 1) continue;
      int ix = tx >> 1;
      if (ix < 0 || ix >= 64) continue;
      const bf16* src = d1 + ((size_t)((n << 6) + iy) * 64 + ix) * 256;
      const float* w0 = wp + ((ky * 3 + kx) * 3) * 256;
      B8 v8; v8.q = *reinterpret_cast<const U4*>(src + lane * 8);
#pragma unroll
      for (int u = 0; u < 8; ++u) {
        int ci = lane * 8 + u;
        float v = (float)v8.h[u];
        a0 = fmaf(v, w0[ci], a0);
        a1 = fmaf(v, w0[256 + ci], a1);
        a2 = fmaf(v, w0[512 + ci], a2);
      }
    }
  }
  for (int off = 16; off; off >>= 1) {
    a0 += __shfl_down(a0, off);
    a1 += __shfl_down(a1, off);
    a2 += __shfl_down(a2, off);
  }
  if (lane == 0) {
    size_t base = (size_t)n * 3 * 16384 + (size_t)oy * 128 + ox;
    out[base]             = tanhf(a0 + bd2[0]);
    out[base + 16384]     = tanhf(a1 + bd2[1]);
    out[base + 2 * 16384] = tanhf(a2 + bd2[2]);
  }
}

// ---------------------------------------------------------------------------
// Weight / codebook repacking
// ---------------------------------------------------------------------------
__global__ void repack_w1_kernel(const float* __restrict__ W1, float* __restrict__ w1p)
{ // [k=(ky*3+kx)*3+ci][co=128] f32 from OIHW(128,3,3,3)
  int i = blockIdx.x * 256 + threadIdx.x;
  if (i >= 3456) return;
  int co = i & 127, k = i >> 7;
  int ci = k % 3, kxy = k / 3, kx = kxy % 3, ky = kxy / 3;
  w1p[i] = W1[((co * 3 + ci) * 3 + ky) * 3 + kx];
}
__global__ void repack_w2_kernel(const float* __restrict__ W2, bf16* __restrict__ w2p)
{ // [co][K=1152] bf16 from OIHW(256,128,3,3)
  int i = blockIdx.x * 256 + threadIdx.x;
  if (i >= 294912) return;
  int co = i / 1152, kk = i % 1152;
  int tap = kk >> 7, ci = kk & 127, ky = tap / 3, kx = tap % 3;
  w2p[i] = (bf16)W2[((co * 128 + ci) * 3 + ky) * 3 + kx];
}
__global__ void repack_w3_kernel(const float* __restrict__ W3, bf16* __restrict__ w3p)
{ // 1x1: OIHW flat == [co][ci]
  int i = blockIdx.x * 256 + threadIdx.x;
  if (i >= 65536) return;
  w3p[i] = (bf16)W3[i];
}
__global__ void repack_wd1_kernel(const float* __restrict__ Wd1, bf16* __restrict__ wd1p)
{ // conv weight[co][ci][ky][kx] = Wd1[ci][co][2-ky][2-kx]; pack [co][K=2304]
  int i = blockIdx.x * 256 + threadIdx.x;
  if (i >= 589824) return;
  int co = i / 2304, kk = i % 2304;
  int tap = kk >> 8, ci = kk & 255, ky = tap / 3, kx = tap % 3;
  wd1p[i] = (bf16)Wd1[((ci * 256 + co) * 3 + (2 - ky)) * 3 + (2 - kx)];
}
__global__ void repack_wd2_kernel(const float* __restrict__ Wd2, float* __restrict__ wd2p)
{ // [tap][co=3][ci=256] f32; weight = Wd2[ci][co][2-ky][2-kx]
  int i = blockIdx.x * 256 + threadIdx.x;
  if (i >= 6912) return;
  int tap = i / 768, r = i % 768, co = r >> 8, ci = r & 255;
  int ky = tap / 3, kx = tap % 3;
  wd2p[i] = Wd2[((ci * 3 + co) * 3 + (2 - ky)) * 3 + (2 - kx)];
}
__global__ void repack_cb_kernel(const float* __restrict__ cb, bf16* __restrict__ cbT)
{ // [code][d] bf16 of -2*codebook (score GEMM "weights")
  int i = blockIdx.x * 256 + threadIdx.x;
  if (i >= 131072) return;
  cbT[i] = (bf16)(-2.f * cb[i]);
}

// ---------------------------------------------------------------------------
#define WSP(T, off) ((T*)(ws + (off)))

extern "C" void kernel_launch(void* const* d_in, const int* in_sizes, int n_in,
                              void* d_out, int out_size, void* d_ws, size_t ws_size,
                              hipStream_t stream)
{
  const float* x    = (const float*)d_in[0];
  const float* W1   = (const float*)d_in[1];
  const float* b1   = (const float*)d_in[2];
  const float* g1   = (const float*)d_in[3];
  const float* be1  = (const float*)d_in[4];
  const float* W2   = (const float*)d_in[5];
  const float* b2   = (const float*)d_in[6];
  const float* g2   = (const float*)d_in[7];
  const float* be2  = (const float*)d_in[8];
  const float* W3   = (const float*)d_in[9];
  const float* b3   = (const float*)d_in[10];
  const float* cb   = (const float*)d_in[11];
  const float* Wd1  = (const float*)d_in[12];
  const float* bd1  = (const float*)d_in[13];
  const float* gd1  = (const float*)d_in[14];
  const float* bed1 = (const float*)d_in[15];
  const float* Wd2  = (const float*)d_in[16];
  const float* bd2  = (const float*)d_in[17];
  float* out = (float*)d_out;
  char* ws = (char*)d_ws;
  (void)in_sizes; (void)n_in; (void)out_size; (void)ws_size;

  // d_out layout: recon(3145728) | loss(1) | z_e(4194304) | z_q(4194304)
  const size_t LOSS_OFF = 3145728, ZE_OFF = 3145729, ZQ_OFF = 7340033;

  size_t cur = 0;
  auto alloc = [&](size_t b) { size_t o = cur; cur += (b + 255) & ~(size_t)255; return o; };
  size_t o_zero = alloc(256);           // zero pad for OOB async A-loads
  size_t o_w1p  = alloc(3456 * 4);
  size_t o_w2p  = alloc(294912 * 2);
  size_t o_w3p  = alloc(65536 * 2);
  size_t o_wd1p = alloc(589824 * 2);
  size_t o_wd2p = alloc(6912 * 4);
  size_t o_cbT  = alloc(131072 * 2);
  size_t o_en   = alloc(512 * 4);
  size_t o_ab   = alloc(512 * 4);
  size_t o_ps   = alloc(1024 * 256 * 4);
  size_t o_pq   = alloc(1024 * 256 * 4);
  size_t o_lp   = alloc(65536 * 4);
  size_t o_idx  = alloc(65536 * 4);
  size_t o_bufA = alloc(268435456ULL);  // h1raw f32 (134MB) -> d1raw f32 (268MB)
  size_t o_bufB = alloc(142737408ULL);  // VQ scores f32 (134MB) -> dilated bf16 (142.7MB)
  size_t o_bufC = alloc(67108864ULL);   // h1 bf16 -> z_e f32 NHWC
  size_t o_bufD = alloc(67108864ULL);   // h2 raw f32
  size_t o_bufE = alloc(33554432ULL);   // h2 bf16 -> z_e bf16
  size_t o_zqbf = alloc(33554432ULL);   // z_q bf16 NHWC
  size_t o_d1bf = alloc(134217728ULL);  // decoder hidden bf16 NHWC

  const bf16* zpad = WSP(const bf16, o_zero);
  (void)hipMemsetAsync(ws + o_zero, 0, 256, stream);

  // --- repack weights / codebook ---
  repack_w1_kernel <<<14,   256, 0, stream>>>(W1,  WSP(float, o_w1p));
  repack_w2_kernel <<<1152, 256, 0, stream>>>(W2,  WSP(bf16,  o_w2p));
  repack_w3_kernel <<<256,  256, 0, stream>>>(W3,  WSP(bf16,  o_w3p));
  repack_wd1_kernel<<<2304, 256, 0, stream>>>(Wd1, WSP(bf16,  o_wd1p));
  repack_wd2_kernel<<<27,   256, 0, stream>>>(Wd2, WSP(float, o_wd2p));
  repack_cb_kernel <<<512,  256, 0, stream>>>(cb,  WSP(bf16,  o_cbT));
  enorm_kernel     <<<512,  32,  0, stream>>>(cb,  WSP(float, o_en));

  // --- encoder conv1 + BN + lrelu ---
  conv1_kernel<<<131072, 256, 0, stream>>>(x, WSP(float, o_w1p), b1, WSP(float, o_bufA));
  bn_stats_kernel<<<1024, 256, 0, stream>>>(WSP(float, o_bufA), 128, 256,
                                            WSP(float, o_ps), WSP(float, o_pq));
  bn_finalize_kernel<<<1, 128, 0, stream>>>(WSP(float, o_ps), WSP(float, o_pq),
                                            1024, 128, 1.f / 262144.f, g1, be1,
                                            WSP(float, o_ab));
  bn_lrelu_bf16_kernel<<<131072, 256, 0, stream>>>(WSP(float, o_bufA), WSP(float, o_ab),
                                                   128, WSP(bf16, o_bufC));

  // --- conv2 (WMMA) + BN + lrelu ---
  conv_gemm_wmma<<<dim3(512, 2), 256, 0, stream>>>(
      WSP(bf16, o_bufC), WSP(bf16, o_w2p), b2, WSP(float, o_bufD), zpad,
      64, 64, 128, 32, 32, 256, 2, 1, 3, 1152);
  bn_stats_kernel<<<1024, 256, 0, stream>>>(WSP(float, o_bufD), 256, 64,
                                            WSP(float, o_ps), WSP(float, o_pq));
  bn_finalize_kernel<<<1, 256, 0, stream>>>(WSP(float, o_ps), WSP(float, o_pq),
                                            1024, 256, 1.f / 65536.f, g2, be2,
                                            WSP(float, o_ab));
  bn_lrelu_bf16_kernel<<<65536, 256, 0, stream>>>(WSP(float, o_bufD), WSP(float, o_ab),
                                                  256, WSP(bf16, o_bufE));

  // --- conv3 1x1 (WMMA) -> z_e ---
  conv_gemm_wmma<<<dim3(512, 2), 256, 0, stream>>>(
      WSP(bf16, o_bufE), WSP(bf16, o_w3p), b3, WSP(float, o_bufC), zpad,
      32, 32, 256, 32, 32, 256, 1, 0, 1, 256);
  ze_out_kernel     <<<65536, 256, 0, stream>>>(WSP(float, o_bufC), out + ZE_OFF);
  f32_to_bf16_kernel<<<65536, 256, 0, stream>>>(WSP(float, o_bufC), WSP(bf16, o_bufE));

  // --- VQ: scores (WMMA GEMM, bias = ||e||^2, weights = -2*codebook), argmin, gather ---
  conv_gemm_wmma<<<dim3(512, 4), 256, 0, stream>>>(
      WSP(bf16, o_bufE), WSP(bf16, o_cbT), WSP(float, o_en), WSP(float, o_bufB), zpad,
      32, 32, 256, 32, 32, 512, 1, 0, 1, 256);
  argmin_kernel<<<8192, 256, 0, stream>>>(WSP(float, o_bufB), WSP(int, o_idx));
  vq_gather_kernel<<<65536, 256, 0, stream>>>(WSP(int, o_idx), cb, WSP(float, o_bufC),
                                              out + ZQ_OFF, WSP(bf16, o_zqbf),
                                              WSP(float, o_lp));
  loss_final_kernel<<<1, 256, 0, stream>>>(WSP(float, o_lp), out + LOSS_OFF);

  // --- decoder convT1 as dense conv over zero-dilated (64,66,66,256) ---
  (void)hipMemsetAsync(ws + o_bufB, 0, 142737408ULL, stream);
  scatter_dil_kernel<<<65536, 256, 0, stream>>>(WSP(bf16, o_zqbf), WSP(bf16, o_bufB));
  conv_gemm_wmma<<<dim3(2048, 2), 256, 0, stream>>>(
      WSP(bf16, o_bufB), WSP(bf16, o_wd1p), bd1, WSP(float, o_bufA), zpad,
      66, 66, 256, 64, 64, 256, 1, 0, 3, 2304);
  bn_stats_kernel<<<1024, 256, 0, stream>>>(WSP(float, o_bufA), 256, 256,
                                            WSP(float, o_ps), WSP(float, o_pq));
  bn_finalize_kernel<<<1, 256, 0, stream>>>(WSP(float, o_ps), WSP(float, o_pq),
                                            1024, 256, 1.f / 262144.f, gd1, bed1,
                                            WSP(float, o_ab));
  bn_lrelu_bf16_kernel<<<262144, 256, 0, stream>>>(WSP(float, o_bufA), WSP(float, o_ab),
                                                   256, WSP(bf16, o_d1bf));

  // --- convT2 + tanh -> recon (NCHW) ---
  deconv2_tanh_kernel<<<131072, 256, 0, stream>>>(WSP(bf16, o_d1bf), WSP(float, o_wd2p),
                                                  bd2, out);
}
